// M5_EncoderDecoder_16449724744091
// MI455X (gfx1250) — matrix operand order repaired and checked by reading, not compiled, not run on hardware
//
#include <hip/hip_runtime.h>
#include <hip/hip_bf16.h>
#include <math.h>

// ---------------------------------------------------------------------------
// M5 EncoderDecoder for MI455X (gfx1250, wave32, WMMA f32_16x16x32_f16)
//
// Shapes: B=4096, T=56, STEPS=28, H=126 (pad 128), 3H=378 (gate-repacked 384),
// ENC=20 (pad 32), DIN=142 (pad 160), ANN=30 (pad 32).
//
// Recurrent kernels keep h AND the GRU weights resident in LDS (dynamic LDS,
// 174 KB enc / 280 KB dec of the WGP's 320 KB); the time loops are barrier-free
// (all state wave-local), so each step is WMMA + ds_load only.
// ---------------------------------------------------------------------------

typedef __attribute__((ext_vector_type(16))) _Float16 v16h;
typedef __attribute__((ext_vector_type(8)))  _Float16 v8h;
typedef __attribute__((ext_vector_type(8)))  float    v8f;

#define B_     4096
#define T_     56
#define STEPS_ 28
#define H_     126
#define HP     128
#define G3     384
#define ENCD   20
#define DECD   15
#define DIN_   142
#define DINP   160
#define ANN_   30

// ---- dynamic LDS layouts (bytes) ------------------------------------------
constexpr int ENC_OFF_WIH = 0;                              // [384][32]  f16
constexpr int ENC_OFF_WHH = ENC_OFF_WIH + G3 * 32 * 2;      // [384][128] f16
constexpr int ENC_OFF_H   = ENC_OFF_WHH + G3 * HP * 2;      // [2][32][128] f16
constexpr int ENC_OFF_X   = ENC_OFF_H + 2 * 32 * HP * 2;    // [32][32]  f16
constexpr int ENC_OFF_R   = ENC_OFF_X + 32 * 32 * 2;        // [32][128] f32
constexpr int ENC_OFF_Z   = ENC_OFF_R + 32 * HP * 4;        // [32][128] f32
constexpr int ENC_SMEM    = ENC_OFF_Z + 32 * HP * 4;        // 174080 B

constexpr int DEC_OFF_WIH = 0;                              // [384][160] f16
constexpr int DEC_OFF_WHH = DEC_OFF_WIH + G3 * DINP * 2;    // [384][128] f16
constexpr int DEC_OFF_H   = DEC_OFF_WHH + G3 * HP * 2;      // [2][32][128] f16
constexpr int DEC_OFF_X   = DEC_OFF_H + 2 * 32 * HP * 2;    // [32][160] f16
constexpr int DEC_OFF_WH  = DEC_OFF_X + 32 * DINP * 2;      // [32][128] f16
constexpr int DEC_OFF_R   = DEC_OFF_WH + 32 * HP * 2;       // [32][128] f16
constexpr int DEC_OFF_Z   = DEC_OFF_R + 32 * HP * 2;        // [32][128] f16
constexpr int DEC_OFF_SC  = DEC_OFF_Z + 32 * HP * 2;        // [32][64]  f32
constexpr int DEC_OFF_PRV = DEC_OFF_SC + 32 * 64 * 4;       // [32]      f32
constexpr int DEC_SMEM    = DEC_OFF_PRV + 32 * 4;           // 280704 B

__device__ __forceinline__ v8f wmma16(v16h a, v16h b, v8f c) {
  // D = A(16x32 f16) * B(32x16 f16) + C(16x16 f32)
  return __builtin_amdgcn_wmma_f32_16x16x32_f16(false, a, false, b, (short)0, c,
                                                false, false);
}

// A-matrix fragment: 16x32 f16 tile, row-major [16][ldk] source (LDS).
// Contiguous k order (same as fragB): lane<16 -> K = k0+0..15 ; lane>=16 -> K = k0+16..31
__device__ __forceinline__ v16h fragA(const _Float16* base, int ldk, int k0, int lane) {
  int m  = lane & 15;
  int kh = (lane < 16) ? 0 : 16;
  const _Float16* p = base + m * ldk + k0 + kh;
  v8h lo = *(const v8h*)(p);
  v8h hi = *(const v8h*)(p + 8);
  v16h r;
#pragma unroll
  for (int i = 0; i < 8; ++i) { r[i] = lo[i]; r[i + 8] = hi[i]; }
  return r;
}

// B-matrix fragment: 32x16 f16 tile from N-major weight array W[n][k] (LDS/global).
// Contiguous k order (same as fragA): lane<16 -> N=lane, K=k0..k0+15 ; lane>=16 -> N=lane-16, K=k0+16..k0+31
__device__ __forceinline__ v16h fragB(const _Float16* W, int ldk,
                                      int ntile, int k0, int lane) {
  int n  = ntile * 16 + (lane & 15);
  int kb = (lane < 16) ? 0 : 16;
  return *(const v16h*)(W + (size_t)n * ldk + k0 + kb);
}

__device__ __forceinline__ float sigm(float x) { return 1.0f / (1.0f + expf(-x)); }

// ---------------------------------------------------------------------------
// Kernel 0: repack all weights to padded, gate-aligned f16 layouts in ws.
// ---------------------------------------------------------------------------
__global__ void prep_kernel(
    const float* __restrict__ s2hW1, const float* __restrict__ s2hW2,
    const float* __restrict__ eWih,  const float* __restrict__ eWhh,
    const float* __restrict__ dWih,  const float* __restrict__ dWhh,
    const float* __restrict__ UW,    const float* __restrict__ WlW,
    const float* __restrict__ ebih,  const float* __restrict__ ebhh,
    const float* __restrict__ dbih,  const float* __restrict__ dbhh,
    const float* __restrict__ Ub,    const float* __restrict__ Wlb,
    const float* __restrict__ VW,    const float* __restrict__ h2oW,
    _Float16* __restrict__ W1g,  _Float16* __restrict__ W2g,
    _Float16* __restrict__ eWihg, _Float16* __restrict__ eWhhg,
    _Float16* __restrict__ dWihg, _Float16* __restrict__ dWhhg,
    _Float16* __restrict__ Ug,    _Float16* __restrict__ Wlg,
    float* __restrict__ biasp) {
  int tid = blockIdx.x * blockDim.x + threadIdx.x;
  int nth = gridDim.x * blockDim.x;
  for (int i = tid; i < 96 * 32; i += nth) {            // W1g [96][32]
    int n = i >> 5, k = i & 31;
    W1g[i] = (_Float16)((k < ANN_) ? s2hW1[n * ANN_ + k] : 0.f);
  }
  for (int i = tid; i < 128 * 96; i += nth) {           // W2g [128][96]
    int n = i / 96, k = i % 96;
    W2g[i] = (_Float16)((n < H_) ? s2hW2[n * 96 + k] : 0.f);
  }
  for (int i = tid; i < G3 * 32; i += nth) {            // eWihg [384][32]
    int n = i >> 5, k = i & 31, g = n >> 7, j = n & 127;
    float v = (j < H_ && k < ENCD) ? eWih[(g * H_ + j) * ENCD + k] : 0.f;
    eWihg[i] = (_Float16)v;
  }
  for (int i = tid; i < G3 * HP; i += nth) {            // eWhhg [384][128]
    int n = i >> 7, k = i & 127, g = n >> 7, j = n & 127;
    float v = (j < H_ && k < H_) ? eWhh[(g * H_ + j) * H_ + k] : 0.f;
    eWhhg[i] = (_Float16)v;
  }
  for (int i = tid; i < G3 * DINP; i += nth) {          // dWihg [384][160]
    int n = i / DINP, k = i % DINP, g = n >> 7, j = n & 127;
    float v = (j < H_ && k < DIN_) ? dWih[(g * H_ + j) * DIN_ + k] : 0.f;
    dWihg[i] = (_Float16)v;
  }
  for (int i = tid; i < G3 * HP; i += nth) {            // dWhhg [384][128]
    int n = i >> 7, k = i & 127, g = n >> 7, j = n & 127;
    float v = (j < H_ && k < H_) ? dWhh[(g * H_ + j) * H_ + k] : 0.f;
    dWhhg[i] = (_Float16)v;
  }
  for (int i = tid; i < HP * HP; i += nth) {            // Ug, Wlg [128][128]
    int n = i >> 7, k = i & 127;
    Ug[i]  = (_Float16)((n < H_ && k < H_) ? UW[n * H_ + k] : 0.f);
    Wlg[i] = (_Float16)((n < H_ && k < H_) ? WlW[n * H_ + k] : 0.f);
  }
  for (int i = tid; i < G3; i += nth) {                 // gate-repacked biases
    int g = i >> 7, j = i & 127;
    biasp[0 * G3 + i] = (j < H_) ? ebih[g * H_ + j] : 0.f;
    biasp[1 * G3 + i] = (j < H_) ? ebhh[g * H_ + j] : 0.f;
    biasp[2 * G3 + i] = (j < H_) ? dbih[g * H_ + j] : 0.f;
    biasp[3 * G3 + i] = (j < H_) ? dbhh[g * H_ + j] : 0.f;
  }
  for (int i = tid; i < HP; i += nth) {                 // padded vectors
    biasp[1536 + i] = (i < H_) ? Ub[i]   : 0.f;
    biasp[1664 + i] = (i < H_) ? Wlb[i]  : 0.f;
    biasp[1792 + i] = (i < H_) ? VW[i]   : 0.f;
    biasp[1920 + i] = (i < H_) ? h2oW[i] : 0.f;
  }
}

// ---------------------------------------------------------------------------
// Kernel 1: MLP  h0 = relu(ann @ W1^T + b1) @ W2^T + b2   -> f32 [4096][128]
// 64 threads (2 waves), 32 batch rows per block, 16 rows per wave.
// ---------------------------------------------------------------------------
__global__ void mlp_kernel(const float* __restrict__ ann,
                           const _Float16* __restrict__ W1g,
                           const float* __restrict__ b1,
                           const _Float16* __restrict__ W2g,
                           const float* __restrict__ b2,
                           float* __restrict__ h0) {
  __shared__ __attribute__((aligned(32))) _Float16 sx[32][32];
  __shared__ __attribute__((aligned(32))) _Float16 sh1[32][96];
  const int lane = threadIdx.x & 31, wv = threadIdx.x >> 5;
  const int rowbase = blockIdx.x * 32, wrow = wv * 16;
  for (int i = lane; i < 16 * 32; i += 32) {
    int m = i >> 5, k = i & 31;
    float v = (k < ANN_) ? ann[(size_t)(rowbase + wrow + m) * ANN_ + k] : 0.f;
    sx[wrow + m][k] = (_Float16)v;
  }
  v16h ax = fragA(&sx[wrow][0], 32, 0, lane);
  const int n = lane & 15, mo = (lane < 16) ? 0 : 8;
  for (int nt = 0; nt < 6; ++nt) {                      // layer 1: N=96
    v8f acc = {};
    acc = wmma16(ax, fragB(W1g, 32, nt, 0, lane), acc);
    int col = nt * 16 + n;
    float bb = b1[col];
#pragma unroll
    for (int i = 0; i < 8; ++i) {
      float v = acc[i] + bb;
      sh1[wrow + mo + i][col] = (_Float16)(v > 0.f ? v : 0.f);
    }
  }
  v16h a1[3];
#pragma unroll
  for (int kc = 0; kc < 3; ++kc) a1[kc] = fragA(&sh1[wrow][0], 96, kc * 32, lane);
  for (int nt = 0; nt < 8; ++nt) {                      // layer 2: N=128
    v8f acc = {};
#pragma unroll
    for (int kc = 0; kc < 3; ++kc)
      acc = wmma16(a1[kc], fragB(W2g, 96, nt, kc * 32, lane), acc);
    int col = nt * 16 + n;
    float bb = (col < H_) ? b2[col] : 0.f;
#pragma unroll
    for (int i = 0; i < 8; ++i)
      h0[(size_t)(rowbase + wrow + mo + i) * HP + col] = acc[i] + bb;
  }
}

// ---------------------------------------------------------------------------
// Kernel 2: encoder GRU. Weights + h live in dynamic LDS (174 KB); the 56-step
// time loop is barrier-free and touches global memory only for x_t / enc_out.
// ---------------------------------------------------------------------------
__global__ void enc_kernel(const float* __restrict__ xdat,
                           const float* __restrict__ h0,
                           const _Float16* __restrict__ Wihg,
                           const _Float16* __restrict__ Whhg,
                           const float* __restrict__ biasp,
                           float* __restrict__ enc_out) {
  extern __shared__ __align__(32) char smem[];
  _Float16* lWih = (_Float16*)(smem + ENC_OFF_WIH);
  _Float16* lWhh = (_Float16*)(smem + ENC_OFF_WHH);
  auto sh_h = (_Float16(*)[32][HP])(smem + ENC_OFF_H);
  auto sh_x = (_Float16(*)[32])(smem + ENC_OFF_X);
  auto sh_r = (float(*)[HP])(smem + ENC_OFF_R);
  auto sh_z = (float(*)[HP])(smem + ENC_OFF_Z);
  const float* bih = biasp + 0 * G3;
  const float* bhh = biasp + 1 * G3;
  const int lane = threadIdx.x & 31, wv = threadIdx.x >> 5;
  const int rowbase = blockIdx.x * 32, wrow = wv * 16;
  // stage weights into LDS once (16B vectors)
  for (int i = threadIdx.x; i < (G3 * 32 * 2) / 16; i += blockDim.x)
    ((uint4*)lWih)[i] = ((const uint4*)Wihg)[i];
  for (int i = threadIdx.x; i < (G3 * HP * 2) / 16; i += blockDim.x)
    ((uint4*)lWhh)[i] = ((const uint4*)Whhg)[i];
  for (int i = threadIdx.x; i < 32 * HP; i += blockDim.x) {
    int m = i >> 7, k = i & 127;
    sh_h[0][m][k] = (_Float16)h0[(size_t)(rowbase + m) * HP + k];
    sh_h[1][m][k] = (_Float16)0.f;
  }
  __syncthreads();
  const int n = lane & 15, mo = (lane < 16) ? 0 : 8;
  int cur = 0;
  for (int t = 0; t < T_; ++t) {
    for (int i = lane; i < 16 * 32; i += 32) {          // stage x_t (wave-local)
      int m = i >> 5, k = i & 31;
      float v = (k < ENCD)
          ? xdat[((size_t)t * B_ + rowbase + wrow + m) * ENCD + k] : 0.f;
      sh_x[wrow + m][k] = (_Float16)v;
    }
    v16h ax = fragA(&sh_x[wrow][0], 32, 0, lane);
    v16h ah[4];
#pragma unroll
    for (int kc = 0; kc < 4; ++kc)
      ah[kc] = fragA(&sh_h[cur][wrow][0], HP, kc * 32, lane);
    for (int nt = 0; nt < 24; ++nt) {
      v8f ai = {}, ag = {};
      ai = wmma16(ax, fragB(lWih, 32, nt, 0, lane), ai);
#pragma unroll
      for (int kc = 0; kc < 4; ++kc)
        ag = wmma16(ah[kc], fragB(lWhh, HP, nt, kc * 32, lane), ag);
      int col = nt * 16 + n;
      float bi = bih[col], bh = bhh[col];
      if (nt < 8) {                                     // r gate
#pragma unroll
        for (int i = 0; i < 8; ++i)
          sh_r[wrow + mo + i][col] = sigm(ai[i] + bi + ag[i] + bh);
      } else if (nt < 16) {                             // z gate
        int c = col - 128;
#pragma unroll
        for (int i = 0; i < 8; ++i)
          sh_z[wrow + mo + i][c] = sigm(ai[i] + bi + ag[i] + bh);
      } else {                                          // n gate + h update
        int c = col - 256;
#pragma unroll
        for (int i = 0; i < 8; ++i) {
          int m = wrow + mo + i;
          float r = sh_r[m][c], z = sh_z[m][c];
          float nn = tanhf(ai[i] + bi + r * (ag[i] + bh));
          float hold = (float)sh_h[cur][m][c];
          float hnew = (1.f - z) * nn + z * hold;
          sh_h[cur ^ 1][m][c] = (_Float16)hnew;
          enc_out[((size_t)t * B_ + rowbase + m) * HP + c] = hnew;
        }
      }
    }
    cur ^= 1;
  }
}

// ---------------------------------------------------------------------------
// Kernel 3: Uo = enc_out @ U^T + Ub, GEMM over R = T*B rows.
// ---------------------------------------------------------------------------
__global__ void uo_kernel(const float* __restrict__ enc_out,
                          const _Float16* __restrict__ Ug,
                          const float* __restrict__ biasp,
                          float* __restrict__ Uo) {
  __shared__ __attribute__((aligned(32))) _Float16 sa[32][HP];
  const float* Ub = biasp + 1536;
  const int lane = threadIdx.x & 31, wv = threadIdx.x >> 5;
  const int rowbase = blockIdx.x * 32, wrow = wv * 16;
  for (int i = lane; i < 16 * HP; i += 32) {
    int m = i >> 7, k = i & 127;
    sa[wrow + m][k] = (_Float16)enc_out[(size_t)(rowbase + wrow + m) * HP + k];
  }
  v16h a[4];
#pragma unroll
  for (int kc = 0; kc < 4; ++kc) a[kc] = fragA(&sa[wrow][0], HP, kc * 32, lane);
  const int n = lane & 15, mo = (lane < 16) ? 0 : 8;
  for (int nt = 0; nt < 8; ++nt) {
    v8f acc = {};
#pragma unroll
    for (int kc = 0; kc < 4; ++kc)
      acc = wmma16(a[kc], fragB(Ug, HP, nt, kc * 32, lane), acc);
    int col = nt * 16 + n;
    float bb = Ub[col];
#pragma unroll
    for (int i = 0; i < 8; ++i)
      Uo[(size_t)(rowbase + wrow + mo + i) * HP + col] = acc[i] + bb;
  }
}

// ---------------------------------------------------------------------------
// Kernel 4: decoder, 28 sequential steps inside the block (barrier-free loop).
// GRU weights resident in dynamic LDS (280 KB); attention streams Uo/enc_out
// from L2 with prefetch hints.
// ---------------------------------------------------------------------------
__global__ void dec_kernel(const float* __restrict__ xdat,   // encoder_data (gt0)
                           const float* __restrict__ ddat,   // decoder_data
                           const float* __restrict__ enc_out,
                           const float* __restrict__ Uo,
                           const _Float16* __restrict__ dWihg,
                           const _Float16* __restrict__ dWhhg,
                           const _Float16* __restrict__ Wlg,
                           const float* __restrict__ biasp,
                           const float* __restrict__ Vb_p,
                           const float* __restrict__ h2ob_p,
                           float* __restrict__ out) {
  extern __shared__ __align__(32) char smem[];
  _Float16* lWih = (_Float16*)(smem + DEC_OFF_WIH);
  _Float16* lWhh = (_Float16*)(smem + DEC_OFF_WHH);
  auto sh_h  = (_Float16(*)[32][HP])(smem + DEC_OFF_H);
  auto sh_x  = (_Float16(*)[DINP])(smem + DEC_OFF_X);
  auto sh_wh = (_Float16(*)[HP])(smem + DEC_OFF_WH);
  auto sh_r  = (_Float16(*)[HP])(smem + DEC_OFF_R);
  auto sh_z  = (_Float16(*)[HP])(smem + DEC_OFF_Z);
  auto sh_sc = (float(*)[64])(smem + DEC_OFF_SC);
  float* sh_prev = (float*)(smem + DEC_OFF_PRV);
  const float* bih = biasp + 2 * G3;
  const float* bhh = biasp + 3 * G3;
  const float* Wlb = biasp + 1664;
  const float* Vw  = biasp + 1792;
  const float* hw  = biasp + 1920;
  const float Vb = Vb_p[0], h2ob = h2ob_p[0];
  const int lane = threadIdx.x & 31, wv = threadIdx.x >> 5;
  const int rowbase = blockIdx.x * 32, wrow = wv * 16;
  // stage weights into LDS once
  for (int i = threadIdx.x; i < (G3 * DINP * 2) / 16; i += blockDim.x)
    ((uint4*)lWih)[i] = ((const uint4*)dWihg)[i];
  for (int i = threadIdx.x; i < (G3 * HP * 2) / 16; i += blockDim.x)
    ((uint4*)lWhh)[i] = ((const uint4*)dWhhg)[i];
  // initial h = final encoder hidden = enc_out[T-1]; prev = encoder_data[-1,:,0]
  for (int i = threadIdx.x; i < 32 * HP; i += blockDim.x) {
    int m = i >> 7, k = i & 127;
    sh_h[0][m][k] =
        (_Float16)enc_out[((size_t)(T_ - 1) * B_ + rowbase + m) * HP + k];
    sh_h[1][m][k] = (_Float16)0.f;
  }
  for (int i = threadIdx.x; i < 32; i += blockDim.x)
    sh_prev[i] = xdat[((size_t)(T_ - 1) * B_ + rowbase + i) * ENCD + 0];
  __syncthreads();
  const int n = lane & 15, mo = (lane < 16) ? 0 : 8;
  const int lm = lane & 15, hh = (lane >> 4) * 64;      // attention lane split
  int cur = 0;
  for (int s = 0; s < STEPS_; ++s) {
    // --- 1. Wh = h @ Wl^T + Wlb -------------------------------------------
    {
      v16h ah[4];
#pragma unroll
      for (int kc = 0; kc < 4; ++kc)
        ah[kc] = fragA(&sh_h[cur][wrow][0], HP, kc * 32, lane);
      for (int nt = 0; nt < 8; ++nt) {
        v8f acc = {};
#pragma unroll
        for (int kc = 0; kc < 4; ++kc)
          acc = wmma16(ah[kc], fragB(Wlg, HP, nt, kc * 32, lane), acc);
        int col = nt * 16 + n;
        float bb = Wlb[col];
#pragma unroll
        for (int i = 0; i < 8; ++i)
          sh_wh[wrow + mo + i][col] = (_Float16)(acc[i] + bb);
      }
    }
    // --- 2. scores[t] = V . tanh(Uo[t] + Wh) + Vb  (2 lanes per row) ------
    {
      int m = wrow + lm, row = rowbase + m;
      for (int t = 0; t < T_; ++t) {
        const float* uop = &Uo[((size_t)t * B_ + row) * HP + hh];
        if (t + 1 < T_)
          __builtin_prefetch(&Uo[((size_t)(t + 1) * B_ + row) * HP + hh], 0, 1);
        float sum = 0.f;
        for (int k = 0; k < 64; ++k)
          sum += Vw[hh + k] * tanhf(uop[k] + (float)sh_wh[m][hh + k]);
        sum += __shfl_xor(sum, 16, 32);
        if (lane < 16) sh_sc[m][t] = sum + Vb;
      }
    }
    // --- 3. softmax over t (one lane per row) ------------------------------
    if (lane < 16) {
      int m = wrow + lane;
      float mx = -3.0e38f;
      for (int t = 0; t < T_; ++t) mx = fmaxf(mx, sh_sc[m][t]);
      float den = 0.f;
      for (int t = 0; t < T_; ++t) {
        float e = expf(sh_sc[m][t] - mx);
        sh_sc[m][t] = e;
        den += e;
      }
      float inv = 1.f / den;
      for (int t = 0; t < T_; ++t) sh_sc[m][t] *= inv;
    }
    // --- 4. attn = sum_t alpha[t]*enc_out[t]; build x = [prev|dec_x|attn] --
    {
      int m = wrow + lm, row = rowbase + m;
      float acc[64];
#pragma unroll
      for (int k = 0; k < 64; ++k) acc[k] = 0.f;
      for (int t = 0; t < T_; ++t) {
        float a = sh_sc[m][t];
        const float* ep = &enc_out[((size_t)t * B_ + row) * HP + hh];
        if (t + 1 < T_)
          __builtin_prefetch(&enc_out[((size_t)(t + 1) * B_ + row) * HP + hh],
                             0, 1);
#pragma unroll 8
        for (int k = 0; k < 64; ++k) acc[k] += a * ep[k];
      }
#pragma unroll
      for (int k = 0; k < 64; ++k) {
        int c = hh + k;
        if (c < H_) sh_x[m][16 + c] = (_Float16)acc[k];
      }
      if (lane < 16) {
        sh_x[m][0] = (_Float16)sh_prev[m];
        for (int k = 0; k < DECD; ++k)
          sh_x[m][1 + k] = (_Float16)ddat[((size_t)s * B_ + row) * DECD + k];
        for (int c = DIN_; c < DINP; ++c) sh_x[m][c] = (_Float16)0.f;
      }
    }
    // --- 5. GRU with decoder weights (from LDS) ----------------------------
    {
      v16h ax[5], ah[4];
#pragma unroll
      for (int kc = 0; kc < 5; ++kc)
        ax[kc] = fragA(&sh_x[wrow][0], DINP, kc * 32, lane);
#pragma unroll
      for (int kc = 0; kc < 4; ++kc)
        ah[kc] = fragA(&sh_h[cur][wrow][0], HP, kc * 32, lane);
      for (int nt = 0; nt < 24; ++nt) {
        v8f ai = {}, ag = {};
#pragma unroll
        for (int kc = 0; kc < 5; ++kc)
          ai = wmma16(ax[kc], fragB(lWih, DINP, nt, kc * 32, lane), ai);
#pragma unroll
        for (int kc = 0; kc < 4; ++kc)
          ag = wmma16(ah[kc], fragB(lWhh, HP, nt, kc * 32, lane), ag);
        int col = nt * 16 + n;
        float bi = bih[col], bh = bhh[col];
        if (nt < 8) {
#pragma unroll
          for (int i = 0; i < 8; ++i)
            sh_r[wrow + mo + i][col] = (_Float16)sigm(ai[i] + bi + ag[i] + bh);
        } else if (nt < 16) {
          int c = col - 128;
#pragma unroll
          for (int i = 0; i < 8; ++i)
            sh_z[wrow + mo + i][c] = (_Float16)sigm(ai[i] + bi + ag[i] + bh);
        } else {
          int c = col - 256;
#pragma unroll
          for (int i = 0; i < 8; ++i) {
            int m = wrow + mo + i;
            float r = (float)sh_r[m][c], z = (float)sh_z[m][c];
            float nn = tanhf(ai[i] + bi + r * (ag[i] + bh));
            float hold = (float)sh_h[cur][m][c];
            sh_h[cur ^ 1][m][c] = (_Float16)((1.f - z) * nn + z * hold);
          }
        }
      }
      cur ^= 1;
    }
    // --- 6. out = h2 @ h2o^T + b ; feed back as prev -----------------------
    {
      int m = wrow + lm;
      float sum = 0.f;
      for (int k = 0; k < 64; ++k)
        sum += hw[hh + k] * (float)sh_h[cur][m][hh + k];
      sum += __shfl_xor(sum, 16, 32);
      if (lane < 16) {
        float o = sum + h2ob;
        out[(size_t)s * B_ + rowbase + m] = o;
        sh_prev[m] = o;
      }
    }
  }
}

// ---------------------------------------------------------------------------
// Host launcher
// ---------------------------------------------------------------------------
extern "C" void kernel_launch(void* const* d_in, const int* in_sizes, int n_in,
                              void* d_out, int out_size, void* d_ws, size_t ws_size,
                              hipStream_t stream) {
  const float* ann   = (const float*)d_in[0];
  const float* encd  = (const float*)d_in[1];
  const float* decd  = (const float*)d_in[2];
  const float* s2hW1 = (const float*)d_in[3];
  const float* s2hb1 = (const float*)d_in[4];
  const float* s2hW2 = (const float*)d_in[5];
  const float* s2hb2 = (const float*)d_in[6];
  const float* eWih  = (const float*)d_in[7];
  const float* eWhh  = (const float*)d_in[8];
  const float* ebih  = (const float*)d_in[9];
  const float* ebhh  = (const float*)d_in[10];
  const float* dWih  = (const float*)d_in[11];
  const float* dWhh  = (const float*)d_in[12];
  const float* dbih  = (const float*)d_in[13];
  const float* dbhh  = (const float*)d_in[14];
  const float* UW    = (const float*)d_in[15];
  const float* Ub    = (const float*)d_in[16];
  const float* WlW   = (const float*)d_in[17];
  const float* Wlb   = (const float*)d_in[18];
  const float* VW    = (const float*)d_in[19];
  const float* Vb    = (const float*)d_in[20];
  const float* h2oW  = (const float*)d_in[21];
  const float* h2ob  = (const float*)d_in[22];
  float* out = (float*)d_out;

  char* ws = (char*)d_ws;
  size_t off = 0;
  auto take = [&](size_t bytes) { char* p = ws + off; off += bytes; return p; };
  float*    h0_ws  = (float*)take((size_t)B_ * HP * 4);
  float*    encout = (float*)take((size_t)T_ * B_ * HP * 4);
  float*    Uo_ws  = (float*)take((size_t)T_ * B_ * HP * 4);
  _Float16* W1g    = (_Float16*)take(96 * 32 * 2);
  _Float16* W2g    = (_Float16*)take(128 * 96 * 2);
  _Float16* eWihg  = (_Float16*)take((size_t)G3 * 32 * 2);
  _Float16* eWhhg  = (_Float16*)take((size_t)G3 * HP * 2);
  _Float16* dWihg  = (_Float16*)take((size_t)G3 * DINP * 2);
  _Float16* dWhhg  = (_Float16*)take((size_t)G3 * HP * 2);
  _Float16* Ug     = (_Float16*)take((size_t)HP * HP * 2);
  _Float16* Wlg    = (_Float16*)take((size_t)HP * HP * 2);
  float*    biasp  = (float*)take(2048 * 4);
  (void)ws_size; (void)n_in; (void)in_sizes; (void)out_size;

  // allow large dynamic LDS (host-side attribute set; graph-capture safe)
  (void)hipFuncSetAttribute((const void*)enc_kernel,
                            hipFuncAttributeMaxDynamicSharedMemorySize,
                            ENC_SMEM);
  (void)hipFuncSetAttribute((const void*)dec_kernel,
                            hipFuncAttributeMaxDynamicSharedMemorySize,
                            DEC_SMEM);

  prep_kernel<<<64, 256, 0, stream>>>(
      s2hW1, s2hW2, eWih, eWhh, dWih, dWhh, UW, WlW,
      ebih, ebhh, dbih, dbhh, Ub, Wlb, VW, h2oW,
      W1g, W2g, eWihg, eWhhg, dWihg, dWhhg, Ug, Wlg, biasp);

  mlp_kernel<<<B_ / 32, 64, 0, stream>>>(ann, W1g, s2hb1, W2g, s2hb2, h0_ws);

  enc_kernel<<<B_ / 32, 64, ENC_SMEM, stream>>>(encd, h0_ws, eWihg, eWhhg,
                                                biasp, encout);

  uo_kernel<<<(T_ * B_) / 32, 64, 0, stream>>>(encout, Ug, biasp, Uo_ws);

  dec_kernel<<<B_ / 32, 64, DEC_SMEM, stream>>>(encd, decd, encout, Uo_ws,
                                                dWihg, dWhhg, Wlg, biasp, Vb,
                                                h2ob, out);
}
